// MessageNEmbedding_9560597201510
// MI455X (gfx1250) — matrix operand order repaired and checked by reading, not compiled, not run on hardware
//
#include <hip/hip_runtime.h>
#include <math.h>

// ---------------------------------------------------------------------------
// CDNA5 (gfx1250) WMMA types
// ---------------------------------------------------------------------------
typedef __attribute__((ext_vector_type(16))) __bf16 v16bf;
typedef __attribute__((ext_vector_type(8)))  float  v8f;

// Native conversion: clang lowers to hardware bf16 converts on gfx1250
// (packed v_cvt where possible), replacing the 3-VALU-op manual RNE sequence.
__device__ __forceinline__ __bf16 f2bf(float f) { return (__bf16)f; }

// A fragment from a pre-resolved row pointer (loop-invariant part hoisted).
// k order used here (contiguous, same as load_b_frag): lane<16 -> K {kb..kb+15}, kb=k0;
// lanes 16..31 same rows, kb=k0+16.
__device__ __forceinline__ v16bf load_a_row(const float* rowp, bool valid, int k0) {
  int lane = (int)(threadIdx.x & 31u);
  int kb   = k0 + ((lane & 16) ? 16 : 0);
  float vals[16];
  if (valid) {
    const float4* p0 = (const float4*)(rowp + kb);
    const float4* p1 = (const float4*)(rowp + kb + 8);
    float4 a0 = p0[0], a1 = p0[1], b0 = p1[0], b1 = p1[1];
    vals[0]=a0.x; vals[1]=a0.y; vals[2]=a0.z; vals[3]=a0.w;
    vals[4]=a1.x; vals[5]=a1.y; vals[6]=a1.z; vals[7]=a1.w;
    vals[8]=b0.x; vals[9]=b0.y; vals[10]=b0.z; vals[11]=b0.w;
    vals[12]=b1.x; vals[13]=b1.y; vals[14]=b1.z; vals[15]=b1.w;
  } else {
#pragma unroll
    for (int i = 0; i < 16; ++i) vals[i] = 0.f;
  }
  v16bf a;
#pragma unroll
  for (int i = 0; i < 16; ++i) a[i] = f2bf(vals[i]);
  return a;
}

// Resolve this lane's A-row pointer once (optionally through rowidx gather).
__device__ __forceinline__ const float* resolve_row(const float* A, long lda,
                                                    int r0, int nrows,
                                                    const int* rowidx, bool& valid) {
  int lane = (int)(threadIdx.x & 31u);
  int r    = r0 + (lane & 15);
  valid    = (r < nrows);
  const float* rowp = A;
  if (valid) {
    long ri = r;
    if (rowidx) { int idx = rowidx[r]; if (idx < 0) valid = false; ri = idx; }
    if (valid) rowp = A + ri * lda;
  }
  return rowp;
}

// B fragment from PRE-CONVERTED bf16 weights: one 32-byte vector load, no
// conversion ops in the hot loop. ldw (elems) and k-steps keep 32B alignment.
// lane<16 -> col=lane, K k0..k0+15 ; lanes 16..31 -> K k0+16..k0+31.
__device__ __forceinline__ v16bf load_b_frag(const __bf16* W, long ldw, int c0, int k0) {
  int lane = (int)(threadIdx.x & 31u);
  int col  = c0 + (lane & 15);
  int kb   = k0 + ((lane & 16) ? 16 : 0);
  return *(const v16bf*)(W + (long)col * ldw + kb);
}

// C/D f32 layout: lane<16 -> col=lane, rows m0..m0+7 (m0=0); lanes>=16 m0=8.
__device__ __forceinline__ void store_d(float* D, long ldd, int r0, int nrows, int c0,
                                        const v8f& acc, const float* bias, int relu) {
  int lane = (int)(threadIdx.x & 31u);
  int col  = c0 + (lane & 15);
  int m0   = (lane & 16) ? 8 : 0;
  float bv = bias ? bias[col] : 0.f;
#pragma unroll
  for (int v = 0; v < 8; ++v) {
    int r = r0 + m0 + v;
    if (r < nrows) {
      float val = acc[v] + bv;
      if (relu) val = fmaxf(val, 0.f);
      D[(long)r * ldd + col] = val;
    }
  }
}

// ---------------------------------------------------------------------------
// Generic GEMM: D[nrows,P] = act( A[nrows,Kd] (opt. row-gathered) @ W[P,Kd]^T + bias )
// Wave = 16 rows x 64 cols (4 accumulators): A fragment loaded once per K step
// and reused for 4 WMMAs -> A redundancy only P/64. W is bf16.
// block = 128 threads (4 waves); wave w -> row tile (blockIdx.x*4 + w).
// grid = (ceil(nrows/64), P/64)
// ---------------------------------------------------------------------------
__global__ void gemm_wmma(const float* A, long lda, const int* rowidx,
                          const __bf16* __restrict__ W, const float* __restrict__ bias,
                          float* D, int nrows, int Kd, int P, int relu) {
  int w  = (int)(threadIdx.x >> 5);
  int r0 = (blockIdx.x * 4 + w) * 16;
  int c0 = blockIdx.y * 64;
  if (r0 >= nrows) return;

  bool valid;
  const float* rowp = resolve_row(A, lda, r0, nrows, rowidx, valid);

  v8f acc[4];
#pragma unroll
  for (int i = 0; i < 4; ++i)
#pragma unroll
    for (int j = 0; j < 8; ++j) acc[i][j] = 0.f;

  for (int k0 = 0; k0 < Kd; k0 += 32) {
    v16bf a = load_a_row(rowp, valid, k0);
#pragma unroll
    for (int ct = 0; ct < 4; ++ct) {
      v16bf b = load_b_frag(W, Kd, c0 + ct * 16, k0);
      acc[ct] = __builtin_amdgcn_wmma_f32_16x16x32_bf16(false, a, false, b,
                                                        (short)0, acc[ct], false, false);
    }
  }
#pragma unroll
  for (int ct = 0; ct < 4; ++ct)
    store_d(D, P, r0, nrows, c0 + ct * 16, acc[ct], bias, relu);
}

// ---------------------------------------------------------------------------
// In-place-safe merge GEMM: D[nrows,128] = relu(concat(A1[.,K1], A2[.,K2]) @ Wm^T + bm)
// One wave owns 16 rows and all 8 col tiles: every load of its rows completes
// before any store -> D may alias A1.
// ---------------------------------------------------------------------------
__global__ void gemm_finalize(const float* A1, int K1,
                              const float* A2, int K2, long ld2,
                              const __bf16* __restrict__ Wm, const float* __restrict__ bias,
                              float* D, int nrows) {
  int w  = (int)(threadIdx.x >> 5);
  int r0 = (blockIdx.x * 4 + w) * 16;
  if (r0 >= nrows) return;

  bool valid1, valid2;
  const float* rowp1 = resolve_row(A1, K1,  r0, nrows, nullptr, valid1);
  const float* rowp2 = resolve_row(A2, ld2, r0, nrows, nullptr, valid2);

  v8f acc[8];
#pragma unroll
  for (int i = 0; i < 8; ++i)
#pragma unroll
    for (int j = 0; j < 8; ++j) acc[i][j] = 0.f;

  int Kd = K1 + K2;
  for (int k0 = 0; k0 < Kd; k0 += 32) {
    v16bf a;
    if (k0 < K1) a = load_a_row(rowp1, valid1, k0);
    else         a = load_a_row(rowp2, valid2, k0 - K1);
#pragma unroll
    for (int ct = 0; ct < 8; ++ct) {
      v16bf b = load_b_frag(Wm, Kd, ct * 16, k0);
      acc[ct] = __builtin_amdgcn_wmma_f32_16x16x32_bf16(false, a, false, b,
                                                        (short)0, acc[ct], false, false);
    }
  }
#pragma unroll
  for (int ct = 0; ct < 8; ++ct)
    store_d(D, 128, r0, nrows, ct * 16, acc[ct], bias, 1);
}

// ---------------------------------------------------------------------------
// Elementwise / segment kernels
// ---------------------------------------------------------------------------
__global__ void cvt_bf16_k(const float* __restrict__ in, __bf16* out, long n) {
  long i = (long)blockIdx.x * blockDim.x + threadIdx.x;
  if (i < n) out[i] = (__bf16)in[i];
}
__global__ void fill_i32_k(int* p, int v, long n) {
  long i = (long)blockIdx.x * blockDim.x + threadIdx.x;
  if (i < n) p[i] = v;
}
__global__ void fill_f32_k(float* p, float v, long n) {
  long i = (long)blockIdx.x * blockDim.x + threadIdx.x;
  if (i < n) p[i] = v;
}
// last incoming edge id per destination (segment_max of eid)
__global__ void seg_last_k(const int* __restrict__ dst, int* last, int E) {
  int e = blockIdx.x * blockDim.x + threadIdx.x;
  if (e < E) atomicMax(&last[dst[e]], e);
}
// per-destination max timestamp (t >= 0, buffer pre-zeroed -> isolated nodes get 0)
__global__ void seg_tmax_k(const int* __restrict__ dst, const float* __restrict__ t,
                           float* t0, int E) {
  int e = blockIdx.x * blockDim.x + threadIdx.x;
  if (e < E) atomicMax((int*)&t0[dst[e]], __float_as_int(t[e]));
}

// GRU gates: s_new = (1-z)*tanh(in + r*hn) + z*h ; h_acc += s_new
__global__ void gru_combine_k(const float* __restrict__ gi, const float* __restrict__ gh,
                              const float* __restrict__ bih, const float* __restrict__ bhh,
                              const float* __restrict__ s_old, float* s_out,
                              float* h_acc, int n) {
  long tid = (long)blockIdx.x * blockDim.x + threadIdx.x;
  long total = (long)n * 128;
  if (tid >= total) return;
  int  c = (int)(tid & 127);
  long r = tid >> 7;
  const float* gir = gi + r * 384;
  const float* ghr = gh + r * 384;
  float ir = gir[c]       + bih[c],       hr = ghr[c]       + bhh[c];
  float iz = gir[c + 128] + bih[c + 128], hz = ghr[c + 128] + bhh[c + 128];
  float in = gir[c + 256] + bih[c + 256], hn = ghr[c + 256] + bhh[c + 256];
  float rg = 1.f / (1.f + __expf(-(ir + hr)));
  float zg = 1.f / (1.f + __expf(-(iz + hz)));
  float ng = tanhf(in + rg * hn);
  float h  = s_old[tid];
  float sn = (1.f - zg) * ng + zg * h;
  if (s_out) s_out[tid] = sn;
  h_acc[tid] += sn;
}

// mailbox scatter: ctx[dst] += [h_src | e | cos((t0[dst]-t)*freq)]  (224 wide)
// one wave per edge, 7 x 32 features
__global__ void ctx_scatter_k(const float* __restrict__ hsrc,
                              const int* __restrict__ src, const int* __restrict__ dst,
                              const float* __restrict__ efeat, const float* __restrict__ t,
                              const float* __restrict__ t0, float* ctx, int E) {
  int wave = blockIdx.x * 8 + (int)(threadIdx.x >> 5);
  int lane = (int)(threadIdx.x & 31u);
  if (wave >= E) return;
  int s = src[wave], d = dst[wave];
  float dt = t0[d] - t[wave];
  const float* hs = hsrc  + (long)s * 128;
  const float* ef = efeat + (long)wave * 32;
  float* cd = ctx + (long)d * 224;
#pragma unroll
  for (int it = 0; it < 7; ++it) {
    int f = it * 32 + lane;
    float v;
    if (f < 128)      v = hs[f];
    else if (f < 160) v = ef[f - 128];
    else {
      float fr = __expf((float)(f - 160) * (-9.f / 63.f) * 2.30258509299f); // 10^(-9i/63)
      v = __cosf(dt * fr);
    }
    atomicAdd(&cd[f], v);
  }
}

// fused MHA weights: Weff = Wo(192x192) @ Wvv(192x224) ; beff = Wo @ bv + bo
__global__ void weff_k(const float* __restrict__ Wo, const float* __restrict__ Wvv,
                       float* Weff) {
  int idx = blockIdx.x * blockDim.x + threadIdx.x;
  if (idx >= 192 * 224) return;
  int i = idx / 224, j = idx % 224;
  float s = 0.f;
  for (int k = 0; k < 192; ++k) s += Wo[i * 192 + k] * Wvv[k * 224 + j];
  Weff[idx] = s;
}
__global__ void beff_k(const float* __restrict__ Wo, const float* __restrict__ bva,
                       const float* __restrict__ bo, float* beff) {
  int i = (int)threadIdx.x;
  if (i < 192) {
    float s = bo[i];
    for (int k = 0; k < 192; ++k) s += Wo[i * 192 + k] * bva[k];
    beff[i] = s;
  }
}

// m_raw output: out[e] = [sa[src] | sb[dst] | cos(t*freq) | e]  (352 wide)
__global__ void mraw_out_k(float* out,
                           const float* __restrict__ sa, const int* __restrict__ src,
                           const float* __restrict__ sb, const int* __restrict__ dst,
                           const float* __restrict__ t, const float* __restrict__ efeat,
                           int E) {
  int wave = blockIdx.x * 8 + (int)(threadIdx.x >> 5);
  int lane = (int)(threadIdx.x & 31u);
  if (wave >= E) return;
  int s = src[wave], d = dst[wave];
  float tv = t[wave];
  const float* A  = sa + (long)s * 128;
  const float* B  = sb + (long)d * 128;
  const float* ef = efeat + (long)wave * 32;
  float* o = out + (long)wave * 352;
#pragma unroll
  for (int it = 0; it < 11; ++it) {
    int c = it * 32 + lane;
    float v;
    if (c < 128)      v = A[c];
    else if (c < 256) v = B[c - 128];
    else if (c < 320) {
      float fr = __expf((float)(c - 256) * (-9.f / 63.f) * 2.30258509299f);
      v = __cosf(tv * fr);
    } else            v = ef[c - 320];
    o[c] = v;
  }
}

// ---------------------------------------------------------------------------
static inline int cdivi(long a, long b) { return (int)((a + b - 1) / b); }

extern "C" void kernel_launch(void* const* d_in, const int* in_sizes, int n_in,
                              void* d_out, int out_size, void* d_ws, size_t ws_size,
                              hipStream_t stream) {
  const float* si_in  = (const float*)d_in[0];
  const float* sj_in  = (const float*)d_in[1];
  const float* sn_in  = (const float*)d_in[2];
  const float* m_raw  = (const float*)d_in[3];
  const float* m_rawr = (const float*)d_in[4];
  const float* efeat  = (const float*)d_in[5];
  const float* t      = (const float*)d_in[6];
  const float* vi     = (const float*)d_in[7];
  const float* vj     = (const float*)d_in[8];
  const float* vn     = (const float*)d_in[9];
  const int* g_src  = (const int*)d_in[10];
  const int* g_dst  = (const int*)d_in[11];
  const int* gr_src = (const int*)d_in[12];
  const int* gr_dst = (const int*)d_in[13];
  const int* gn_src = (const int*)d_in[14];
  const int* gn_dst = (const int*)d_in[15];
  const float* Wih = (const float*)d_in[16];
  const float* Whh = (const float*)d_in[17];
  const float* bih = (const float*)d_in[18];
  const float* bhh = (const float*)d_in[19];
  const float* Wu  = (const float*)d_in[20];
  const float* bu  = (const float*)d_in[21];
  const float* Wv  = (const float*)d_in[22];
  const float* bvp = (const float*)d_in[23];
  const float* Wm  = (const float*)d_in[24];
  const float* bm  = (const float*)d_in[25];
  const float* Wvv = (const float*)d_in[28];
  const float* b_in= (const float*)d_in[29];
  const float* Wo  = (const float*)d_in[30];
  const float* bo  = (const float*)d_in[31];

  const int NU = in_sizes[0] / 128;
  const int NI = in_sizes[1] / 128;
  const int E  = in_sizes[6];
  const long maxN = (NU > NI ? NU : NI);

  float* out  = (float*)d_out;
  float* o_hi = out;
  float* o_hj = o_hi + (long)NU * 128;
  float* o_hn = o_hj + (long)NI * 128;
  float* o_si = o_hn + (long)NI * 128;
  float* o_sj = o_si + (long)NU * 128;
  float* o_mi = o_sj + (long)NI * 128;      // E*352 region -> scratch until phase C
  float* o_mj = o_mi + (long)E * 352;       // E*352 region -> scratch until phase C

  // large scratch aliased into m_raw output regions
  float* gi_buf = o_mi;                     // [maxN,384]
  float* gh_buf = o_mj;                     // [maxN,384]
  float* ctxb   = o_mi;                     // [maxN,224]
  float* ht_i   = o_mi + maxN * 224;        // [NU,192]
  float* ht_tmp = o_mj;                     // [maxN,192]

  // small scratch in d_ws (offsets keep the bf16 pool 32B-aligned)
  long Na = (maxN + 7) & ~7L;
  int*    last = (int*)d_ws;                // [Na] i32
  float*  t0   = (float*)d_ws + Na;         // [Na] f32
  float*  Weff = t0 + Na;                   // [192*224] f32
  float*  beff = Weff + 192 * 224;          // [192] f32
  __bf16* bWih = (__bf16*)(beff + 192);     // 384*352
  __bf16* bWhh = bWih + 384 * 352;          // 384*128
  __bf16* bWu  = bWhh + 384 * 128;          // 128*128
  __bf16* bWv  = bWu  + 128 * 128;          // 128*128
  __bf16* bWm  = bWv  + 128 * 128;          // 128*320
  __bf16* bWeff= bWm  + 128 * 320;          // 192*224

  dim3 blk256(256), blk128(128);

  // ---- Phase 0: weight prep (bf16 pre-conversion + fused MHA weights) --
  cvt_bf16_k<<<cdivi(384 * 352, 256), blk256, 0, stream>>>(Wih, bWih, 384 * 352);
  cvt_bf16_k<<<cdivi(384 * 128, 256), blk256, 0, stream>>>(Whh, bWhh, 384 * 128);
  cvt_bf16_k<<<cdivi(128 * 128, 256), blk256, 0, stream>>>(Wu,  bWu,  128 * 128);
  cvt_bf16_k<<<cdivi(128 * 128, 256), blk256, 0, stream>>>(Wv,  bWv,  128 * 128);
  cvt_bf16_k<<<cdivi(128 * 320, 256), blk256, 0, stream>>>(Wm,  bWm,  128 * 320);
  weff_k<<<cdivi(192 * 224, 256), blk256, 0, stream>>>(Wo, Wvv, Weff);
  beff_k<<<1, 192, 0, stream>>>(Wo, b_in + 384, bo, beff);
  cvt_bf16_k<<<cdivi(192 * 224, 256), blk256, 0, stream>>>(Weff, bWeff, 192 * 224);

  auto gemm = [&](const float* A, long lda, const int* ridx, const __bf16* W,
                  const float* bias, float* D, int nrows, int Kd, int P, int relu) {
    dim3 grid(cdivi(nrows, 64), P / 64);
    gemm_wmma<<<grid, blk128, 0, stream>>>(A, lda, ridx, W, bias, D, nrows, Kd, P, relu);
  };

  // ---- Phase A: GRU memory update + pre-attention h --------------------
  // set i (users): x = last msg of gr over m_raw_r
  fill_i32_k<<<cdivi(NU, 256), blk256, 0, stream>>>(last, -1, NU);
  seg_last_k<<<cdivi(E, 256), blk256, 0, stream>>>(gr_dst, last, E);
  gemm(m_rawr, 352, last, bWih, nullptr, gi_buf, NU, 352, 384, 0);
  gemm(si_in, 128, nullptr, bWhh, nullptr, gh_buf, NU, 128, 384, 0);
  gemm(vi, 128, nullptr, bWu, bu, o_hi, NU, 128, 128, 1);
  gru_combine_k<<<cdivi((long)NU * 128, 256), blk256, 0, stream>>>(
      gi_buf, gh_buf, bih, bhh, si_in, o_si, o_hi, NU);

  // set j (items): x = last msg of g over m_raw
  fill_i32_k<<<cdivi(NI, 256), blk256, 0, stream>>>(last, -1, NI);
  seg_last_k<<<cdivi(E, 256), blk256, 0, stream>>>(g_dst, last, E);
  gemm(m_raw, 352, last, bWih, nullptr, gi_buf, NI, 352, 384, 0);
  gemm(sj_in, 128, nullptr, bWhh, nullptr, gh_buf, NI, 128, 384, 0);
  gemm(vj, 128, nullptr, bWv, bvp, o_hj, NI, 128, 128, 1);
  gru_combine_k<<<cdivi((long)NI * 128, 256), blk256, 0, stream>>>(
      gi_buf, gh_buf, bih, bhh, sj_in, o_sj, o_hj, NI);

  // set n (neg items): x = last msg of gn over m_raw; s_new not an output
  fill_i32_k<<<cdivi(NI, 256), blk256, 0, stream>>>(last, -1, NI);
  seg_last_k<<<cdivi(E, 256), blk256, 0, stream>>>(gn_dst, last, E);
  gemm(m_raw, 352, last, bWih, nullptr, gi_buf, NI, 352, 384, 0);
  gemm(sn_in, 128, nullptr, bWhh, nullptr, gh_buf, NI, 128, 384, 0);
  gemm(vn, 128, nullptr, bWv, bvp, o_hn, NI, 128, 128, 1);
  gru_combine_k<<<cdivi((long)NI * 128, 256), blk256, 0, stream>>>(
      gi_buf, gh_buf, bih, bhh, sn_in, nullptr, o_hn, NI);

  // ---- Phase B: mailbox attention + merge ------------------------------
  // ci (graph gr, h_src = h_j, N_U destinations) -> ht_i (kept, finalize hi last)
  fill_f32_k<<<cdivi(NU, 256), blk256, 0, stream>>>(t0, 0.f, NU);
  seg_tmax_k<<<cdivi(E, 256), blk256, 0, stream>>>(gr_dst, t, t0, E);
  fill_f32_k<<<cdivi((long)NU * 224, 256), blk256, 0, stream>>>(ctxb, 0.f, (long)NU * 224);
  ctx_scatter_k<<<cdivi(E, 8), blk256, 0, stream>>>(o_hj, gr_src, gr_dst, efeat, t, t0, ctxb, E);
  gemm(ctxb, 224, nullptr, bWeff, beff, ht_i, NU, 224, 192, 0);

  // cj (graph g, h_src = h_i) -> finalize hj in place
  fill_f32_k<<<cdivi(NI, 256), blk256, 0, stream>>>(t0, 0.f, NI);
  seg_tmax_k<<<cdivi(E, 256), blk256, 0, stream>>>(g_dst, t, t0, E);
  fill_f32_k<<<cdivi((long)NI * 224, 256), blk256, 0, stream>>>(ctxb, 0.f, (long)NI * 224);
  ctx_scatter_k<<<cdivi(E, 8), blk256, 0, stream>>>(o_hi, g_src, g_dst, efeat, t, t0, ctxb, E);
  gemm(ctxb, 224, nullptr, bWeff, beff, ht_tmp, NI, 224, 192, 0);
  gemm_finalize<<<cdivi(NI, 64), blk128, 0, stream>>>(o_hj, 128, ht_tmp, 192, 192, bWm, bm, o_hj, NI);

  // cn (graph gn, h_src = h_i) -> finalize hn in place
  fill_f32_k<<<cdivi(NI, 256), blk256, 0, stream>>>(t0, 0.f, NI);
  seg_tmax_k<<<cdivi(E, 256), blk256, 0, stream>>>(gn_dst, t, t0, E);
  fill_f32_k<<<cdivi((long)NI * 224, 256), blk256, 0, stream>>>(ctxb, 0.f, (long)NI * 224);
  ctx_scatter_k<<<cdivi(E, 8), blk256, 0, stream>>>(o_hi, gn_src, gn_dst, efeat, t, t0, ctxb, E);
  gemm(ctxb, 224, nullptr, bWeff, beff, ht_tmp, NI, 224, 192, 0);
  gemm_finalize<<<cdivi(NI, 64), blk128, 0, stream>>>(o_hn, 128, ht_tmp, 192, 192, bWm, bm, o_hn, NI);

  // finalize hi (h_i no longer needed as attention source)
  gemm_finalize<<<cdivi(NU, 64), blk128, 0, stream>>>(o_hi, 128, ht_i, 192, 192, bWm, bm, o_hi, NU);

  // ---- Phase C: edge message outputs (overwrite the scratch regions) ---
  mraw_out_k<<<cdivi(E, 8), blk256, 0, stream>>>(o_mi, o_si, g_src, o_sj, g_dst, t, efeat, E);
  mraw_out_k<<<cdivi(E, 8), blk256, 0, stream>>>(o_mj, o_sj, gr_src, o_si, gr_dst, t, efeat, E);

  (void)n_in; (void)out_size; (void)ws_size;
}
